// AttentiveStatsPool_71098888618146
// MI455X (gfx1250) — compile-verified, matrix operand order checked by reading, not run on hardware
//
#include <hip/hip_runtime.h>
#include <hip/hip_bf16.h>
#include <math.h>

// ---- problem constants (from reference setup_inputs) ----
#define B_  16
#define C_  1536
#define T_  2000
#define H_  128
#define EPS 1e-5f

typedef __bf16 bf16_t;
typedef __attribute__((ext_vector_type(16))) __bf16 v16bf;
typedef __attribute__((ext_vector_type(8)))  __bf16 v8bf;
typedef __attribute__((ext_vector_type(8)))  float  v8f;

// Combine two 16-byte LDS loads into one WMMA fragment.
__device__ __forceinline__ v16bf ld2frag(const bf16_t* p0, const bf16_t* p1) {
  v8bf lo = *(const v8bf*)p0;
  v8bf hi = *(const v8bf*)p1;
  return __builtin_shufflevector(lo, hi, 0, 1, 2, 3, 4, 5, 6, 7,
                                 8, 9, 10, 11, 12, 13, 14, 15);
}

// Fragment-major index for the B tile (32k x 64n), lane-row padded to 24.
// ISA 7.12.2 16-bit operand layout: lane = half*16 + (n&15), half = which
// 8-chunk of k; within a lane, p = (k&7) + 8*((k>>4)&1) is contiguous.
__device__ __forceinline__ int bidx(int k, int n) {
  int j = n >> 4, l15 = n & 15;
  int half = (k >> 3) & 1;
  int p = (k & 7) | (((k >> 4) & 1) << 3);
  return (((j << 5) | (half << 4) | l15) * 24) + p;
}

// ---- K0: convert w1[:, :C] and w2 to bf16 ----
__global__ void k_convw(const float* __restrict__ w1, const float* __restrict__ w2,
                        bf16_t* __restrict__ w1b, bf16_t* __restrict__ w2b) {
  int i = blockIdx.x * 256 + threadIdx.x;
  const int n = H_ * C_;
  if (i < n) {
    int h = i / C_, c = i % C_;
    w1b[i] = (bf16_t)w1[(size_t)h * (3 * C_) + c];
  } else if (i < 2 * n) {
    int j = i - n;
    w2b[j] = (bf16_t)w2[j];
  }
}

// ---- K1: masked global mean/std per (b,c) over T; also emit x in bf16 ----
__global__ void k_stats0(const float* __restrict__ x, const float* __restrict__ mask,
                         bf16_t* __restrict__ xb, float* __restrict__ mean0,
                         float* __restrict__ std0) {
  int c = blockIdx.x, b = blockIdx.y, tid = threadIdx.x;
  const float* xr = x + ((size_t)(b * C_ + c)) * T_;
  const float* mr = mask + (size_t)b * T_;
  bf16_t* xbr = xb + ((size_t)(b * C_ + c)) * T_;
  float sm = 0.f, sx = 0.f, sx2 = 0.f;
  for (int t = tid; t < T_; t += 256) {
    float m = mr[t], v = xr[t];
    xbr[t] = (bf16_t)v;
    sm += m; sx += m * v; sx2 += m * v * v;
  }
  __shared__ float r0[256], r1[256], r2[256];
  r0[tid] = sm; r1[tid] = sx; r2[tid] = sx2;
  __syncthreads();
  for (int s = 128; s > 0; s >>= 1) {
    if (tid < s) { r0[tid] += r0[tid + s]; r1[tid] += r1[tid + s]; r2[tid] += r2[tid + s]; }
    __syncthreads();
  }
  if (tid == 0) {
    float inv = 1.0f / r0[0];
    float mu = r1[0] * inv, e2 = r2[0] * inv;
    mean0[b * C_ + c] = mu;
    std0[b * C_ + c] = sqrtf(fmaxf(e2 - mu * mu, EPS));
  }
}

// ---- K2: fold mean0/std0 columns of w1 + b1 into per-batch bias c1 ----
__global__ void k_c1(const float* __restrict__ w1, const float* __restrict__ b1,
                     const float* __restrict__ mean0, const float* __restrict__ std0,
                     float* __restrict__ c1) {
  int h = blockIdx.x, b = blockIdx.y, tid = threadIdx.x;
  const float* wm = w1 + (size_t)h * (3 * C_) + C_;
  const float* ws = w1 + (size_t)h * (3 * C_) + 2 * C_;
  float s = 0.f;
  for (int c = tid; c < C_; c += 256)
    s += wm[c] * mean0[b * C_ + c] + ws[c] * std0[b * C_ + c];
  __shared__ float r[256];
  r[tid] = s;
  __syncthreads();
  for (int st = 128; st > 0; st >>= 1) { if (tid < st) r[tid] += r[tid + st]; __syncthreads(); }
  if (tid == 0) c1[b * H_ + h] = b1[h] + r[0];
}

// ---- K3: GEMM1  h(128 x T) = w1b(128 x C) * xb(C x T) + c1, relu, LN(H), tanh ----
__global__ __launch_bounds__(256) void k_gemm1(const bf16_t* __restrict__ w1b,
                                               const bf16_t* __restrict__ xb,
                                               const float* __restrict__ c1,
                                               const float* __restrict__ g1,
                                               const float* __restrict__ be1,
                                               bf16_t* __restrict__ hb) {
  __shared__ __align__(16) union {
    struct {
      bf16_t As[128][40];       // 32 used, padded to 40 (80B rows)
      bf16_t BsP[4 * 32 * 24];  // fragment-major, 24-elem lane rows
    } st;
    float Cs[128][65];
  } sm;
  int tid = threadIdx.x;
  int wv = tid >> 5, lane = tid & 31;  // wave32
  int half = lane >> 4, l15 = lane & 15, hk = half * 8;
  int t0 = blockIdx.x * 64, b = blockIdx.y;
  v8f acc[4] = {};
  for (int kk = 0; kk < C_; kk += 32) {
    if (kk + 32 < C_)
      __builtin_prefetch(&xb[((size_t)(b * C_ + kk + 32)) * T_ + t0], 0, 0);
    // stage A: 128x32, 16B vector copies (2 per thread)
    for (int i = tid; i < 128 * 4; i += 256) {
      int m = i >> 2, ch = (i & 3) * 8;
      *(v8bf*)&sm.st.As[m][ch] = *(const v8bf*)&w1b[(size_t)m * C_ + kk + ch];
    }
    // stage B: 32x64, 16B global loads, fragment-order LDS scatter (1/thread)
    for (int i = tid; i < 32 * 8; i += 256) {
      int k = i >> 3, nc = (i & 7) * 8;
      int t = t0 + nc;
      v8bf v;
      if (t + 8 <= T_) {
        v = *(const v8bf*)&xb[((size_t)(b * C_ + kk + k)) * T_ + t];
      } else {
#pragma unroll
        for (int e = 0; e < 8; ++e)
          v[e] = (t + e < T_) ? xb[((size_t)(b * C_ + kk + k)) * T_ + t + e]
                              : (bf16_t)0.0f;
      }
#pragma unroll
      for (int e = 0; e < 8; ++e) sm.st.BsP[bidx(k, nc + e)] = v[e];
    }
    __syncthreads();
    const bf16_t* pa = &sm.st.As[16 * wv + l15][hk];
    v16bf a = ld2frag(pa, pa + 16);
#pragma unroll
    for (int j = 0; j < 4; ++j) {
      const bf16_t* pb = &sm.st.BsP[((j << 5) | lane) * 24];
      v16bf bb = ld2frag(pb, pb + 8);
      acc[j] = __builtin_amdgcn_wmma_f32_16x16x32_bf16(false, a, false, bb,
                                                       (short)0, acc[j], false, false);
    }
    __syncthreads();
  }
  // epilogue: + c1, relu -> LDS tile (full H=128 per column)
#pragma unroll
  for (int j = 0; j < 4; ++j) {
#pragma unroll
    for (int v = 0; v < 8; ++v) {
      int m = 16 * wv + v + 8 * half;
      int n = 16 * j + l15;
      float val = acc[j][v] + c1[b * H_ + m];
      sm.Cs[m][n] = fmaxf(val, 0.0f);
    }
  }
  __syncthreads();
  // LayerNorm over H per column, then tanh
  if (tid < 64 && t0 + tid < T_) {
    int n = tid;
    float mu = 0.f;
    for (int m = 0; m < 128; ++m) mu += sm.Cs[m][n];
    mu *= (1.0f / 128.0f);
    float var = 0.f;
    for (int m = 0; m < 128; ++m) { float d = sm.Cs[m][n] - mu; var += d * d; }
    var *= (1.0f / 128.0f);
    float r = rsqrtf(var + EPS);
    for (int m = 0; m < 128; ++m) {
      float y = (sm.Cs[m][n] - mu) * r * g1[m] + be1[m];
      sm.Cs[m][n] = tanhf(y);
    }
  }
  __syncthreads();
  // vectorized bf16 store of h
  for (int i = tid; i < 128 * 8; i += 256) {
    int m = i >> 3, ch = (i & 7) * 8;
    int t = t0 + ch;
    if (t + 8 <= T_) {
      v8bf v;
#pragma unroll
      for (int e = 0; e < 8; ++e) v[e] = (bf16_t)sm.Cs[m][ch + e];
      *(v8bf*)&hb[((size_t)(b * H_ + m)) * T_ + t] = v;
    } else {
      for (int e = 0; e < 8 && t + e < T_; ++e)
        hb[((size_t)(b * H_ + m)) * T_ + t + e] = (bf16_t)sm.Cs[m][ch + e];
    }
  }
}

// ---- K4: GEMM2  logits(C x T) = w2b(C x H) * hb(H x T) + b2 ----
__global__ __launch_bounds__(256) void k_gemm2(const bf16_t* __restrict__ w2b,
                                               const bf16_t* __restrict__ hb,
                                               const float* __restrict__ b2,
                                               float* __restrict__ lg) {
  __shared__ __align__(16) struct {
    bf16_t As[128][40];
    bf16_t BsP[4 * 32 * 24];
  } st;
  int tid = threadIdx.x;
  int wv = tid >> 5, lane = tid & 31;
  int half = lane >> 4, l15 = lane & 15, hk = half * 8;
  int t0 = blockIdx.x * 64, m0 = blockIdx.y * 128, b = blockIdx.z;
  v8f acc[4] = {};
  for (int kk = 0; kk < H_; kk += 32) {
    if (kk + 32 < H_)
      __builtin_prefetch(&hb[((size_t)(b * H_ + kk + 32)) * T_ + t0], 0, 0);
    for (int i = tid; i < 128 * 4; i += 256) {
      int m = i >> 2, ch = (i & 3) * 8;
      *(v8bf*)&st.As[m][ch] = *(const v8bf*)&w2b[(size_t)(m0 + m) * H_ + kk + ch];
    }
    for (int i = tid; i < 32 * 8; i += 256) {
      int k = i >> 3, nc = (i & 7) * 8;
      int t = t0 + nc;
      v8bf v;
      if (t + 8 <= T_) {
        v = *(const v8bf*)&hb[((size_t)(b * H_ + kk + k)) * T_ + t];
      } else {
#pragma unroll
        for (int e = 0; e < 8; ++e)
          v[e] = (t + e < T_) ? hb[((size_t)(b * H_ + kk + k)) * T_ + t + e]
                              : (bf16_t)0.0f;
      }
#pragma unroll
      for (int e = 0; e < 8; ++e) st.BsP[bidx(k, nc + e)] = v[e];
    }
    __syncthreads();
    const bf16_t* pa = &st.As[16 * wv + l15][hk];
    v16bf a = ld2frag(pa, pa + 16);
#pragma unroll
    for (int j = 0; j < 4; ++j) {
      const bf16_t* pb = &st.BsP[((j << 5) | lane) * 24];
      v16bf bb = ld2frag(pb, pb + 8);
      acc[j] = __builtin_amdgcn_wmma_f32_16x16x32_bf16(false, a, false, bb,
                                                       (short)0, acc[j], false, false);
    }
    __syncthreads();
  }
#pragma unroll
  for (int j = 0; j < 4; ++j) {
#pragma unroll
    for (int v = 0; v < 8; ++v) {
      int m = m0 + 16 * wv + v + 8 * half;
      int n = 16 * j + l15, t = t0 + n;
      if (t < T_) lg[((size_t)(b * C_ + m)) * T_ + t] = acc[j][v] + b2[m];
    }
  }
}

// ---- K5: masked softmax over T + attention-weighted mean/std, fused ----
__global__ void k_softmax_stats(const float* __restrict__ lg, const float* __restrict__ x,
                                const float* __restrict__ mask, float* __restrict__ pooled) {
  int c = blockIdx.x, b = blockIdx.y, tid = threadIdx.x;
  const float* lr = lg + ((size_t)(b * C_ + c)) * T_;
  const float* xr = x + ((size_t)(b * C_ + c)) * T_;
  const float* mr = mask + (size_t)b * T_;
  __shared__ float r[256], r1[256], r2[256];
  float mx = -INFINITY;
  for (int t = tid; t < T_; t += 256)
    if (mr[t] != 0.0f) mx = fmaxf(mx, lr[t]);
  r[tid] = mx;
  __syncthreads();
  for (int s = 128; s > 0; s >>= 1) {
    if (tid < s) r[tid] = fmaxf(r[tid], r[tid + s]);
    __syncthreads();
  }
  mx = r[0];
  __syncthreads();
  float se = 0.f, sx = 0.f, sx2 = 0.f;
  for (int t = tid; t < T_; t += 256) {
    if (mr[t] != 0.0f) {
      float e = expf(lr[t] - mx);
      float v = xr[t];
      se += e; sx += e * v; sx2 += e * v * v;
    }
  }
  r[tid] = se; r1[tid] = sx; r2[tid] = sx2;
  __syncthreads();
  for (int s = 128; s > 0; s >>= 1) {
    if (tid < s) { r[tid] += r[tid + s]; r1[tid] += r1[tid + s]; r2[tid] += r2[tid + s]; }
    __syncthreads();
  }
  if (tid == 0) {
    float inv = 1.0f / r[0];
    float mu = r1[0] * inv, e2 = r2[0] * inv;
    pooled[(size_t)b * (2 * C_) + c] = mu;
    pooled[(size_t)b * (2 * C_) + C_ + c] = sqrtf(fmaxf(e2 - mu * mu, EPS));
  }
}

// ---- K6: final LayerNorm over 2C per batch -> d_out ----
__global__ void k_lnout(const float* __restrict__ pooled, const float* __restrict__ g2,
                        const float* __restrict__ be2, float* __restrict__ out) {
  int b = blockIdx.x, tid = threadIdx.x;
  const int N = 2 * C_;
  const float* pr = pooled + (size_t)b * N;
  __shared__ float r[256];
  float s = 0.f;
  for (int i = tid; i < N; i += 256) s += pr[i];
  r[tid] = s;
  __syncthreads();
  for (int st = 128; st > 0; st >>= 1) { if (tid < st) r[tid] += r[tid + st]; __syncthreads(); }
  float mu = r[0] / (float)N;
  __syncthreads();
  float v = 0.f;
  for (int i = tid; i < N; i += 256) { float d = pr[i] - mu; v += d * d; }
  r[tid] = v;
  __syncthreads();
  for (int st = 128; st > 0; st >>= 1) { if (tid < st) r[tid] += r[tid + st]; __syncthreads(); }
  float rs = rsqrtf(r[0] / (float)N + EPS);
  for (int i = tid; i < N; i += 256)
    out[(size_t)b * N + i] = (pr[i] - mu) * rs * g2[i] + be2[i];
}

extern "C" void kernel_launch(void* const* d_in, const int* in_sizes, int n_in,
                              void* d_out, int out_size, void* d_ws, size_t ws_size,
                              hipStream_t stream) {
  const float* x    = (const float*)d_in[0];
  const float* mask = (const float*)d_in[1];
  const float* w1   = (const float*)d_in[2];
  const float* b1   = (const float*)d_in[3];
  const float* g1   = (const float*)d_in[4];
  const float* be1  = (const float*)d_in[5];
  const float* w2   = (const float*)d_in[6];
  const float* b2   = (const float*)d_in[7];
  const float* g2   = (const float*)d_in[8];
  const float* be2  = (const float*)d_in[9];
  float* out = (float*)d_out;

  char* ws = (char*)d_ws;
  size_t o = 0;
  auto alloc = [&](size_t bytes) {
    size_t p = o;
    o += (bytes + 255) & ~(size_t)255;
    return p;
  };
  bf16_t* xb     = (bf16_t*)(ws + alloc((size_t)B_ * C_ * T_ * 2));
  bf16_t* hb     = (bf16_t*)(ws + alloc((size_t)B_ * H_ * T_ * 2));
  float*  lgt    = (float*)(ws + alloc((size_t)B_ * C_ * T_ * 4));
  float*  mean0  = (float*)(ws + alloc((size_t)B_ * C_ * 4));
  float*  std0   = (float*)(ws + alloc((size_t)B_ * C_ * 4));
  float*  c1     = (float*)(ws + alloc((size_t)B_ * H_ * 4));
  float*  pooled = (float*)(ws + alloc((size_t)B_ * 2 * C_ * 4));
  bf16_t* w1b    = (bf16_t*)(ws + alloc((size_t)H_ * C_ * 2));
  bf16_t* w2b    = (bf16_t*)(ws + alloc((size_t)C_ * H_ * 2));

  k_convw<<<(2 * H_ * C_ + 255) / 256, 256, 0, stream>>>(w1, w2, w1b, w2b);
  k_stats0<<<dim3(C_, B_), 256, 0, stream>>>(x, mask, xb, mean0, std0);
  k_c1<<<dim3(H_, B_), 256, 0, stream>>>(w1, b1, mean0, std0, c1);
  k_gemm1<<<dim3((T_ + 63) / 64, B_), 256, 0, stream>>>(w1b, xb, c1, g1, be1, hb);
  k_gemm2<<<dim3((T_ + 63) / 64, C_ / 128, B_), 256, 0, stream>>>(w2b, hb, b2, lgt);
  k_softmax_stats<<<dim3(C_, B_), 256, 0, stream>>>(lgt, x, mask, pooled);
  k_lnout<<<B_, 256, 0, stream>>>(pooled, g2, be2, out);
}
